// SRGModel_16432544875093
// MI455X (gfx1250) — matrix operand order repaired and checked by reading, not compiled, not run on hardware
//
#include <hip/hip_runtime.h>
#include <hip/hip_bf16.h>
#include <math.h>

// ---------------------------------------------------------------------------
// Problem constants: B=32, W=12, N=128, H=128, channel=2
// ---------------------------------------------------------------------------
#define BB 32
#define WW 12
#define NN 128
#define HH 128
#define BW (BB * WW)               // 384
#define NODES (2 * NN)             // 256 rows per window
#define PIX (NN * NN)              // 16384
#define GE_ELEMS ((size_t)BW * NODES * NN)   // 12,582,912
#define IN_FEAT (NODES * HH)       // 32768  (GRU input size)
#define GATES (3 * HH)             // 384
#define OUT_FFN (NN * NN)          // 16384
#define LOSS_CNT ((float)GE_ELEMS)
#define TILE_HALFS 512             // one packed 32x16 B tile = 32 lanes * 16 halves
#define NTI (GATES / 16)           // 24 ntiles of the gate dimension
#define KTI (IN_FEAT / 32)         // 1024 k-tiles of the GRU input dim

typedef _Float16 half_t;
typedef __attribute__((ext_vector_type(16))) _Float16 v16h;
typedef __attribute__((ext_vector_type(8)))  _Float16 v8h;
typedef __attribute__((ext_vector_type(8)))  float    v8f;
typedef __attribute__((ext_vector_type(4)))  int      v4i;

// ---------------------------------------------------------------------------
// Workspace layout (bytes, 256-aligned). Packed-B layouts are permutations of
// the plain K x N matrices, so sizes are unchanged.
// ---------------------------------------------------------------------------
#define OFF_HGE   ((size_t)0)                        // f16, GE_ELEMS            (25,165,824 B)
#define OFF_TE    ((size_t)25165824)                 // f16, GE_ELEMS (=X)       (25,165,824 B)
#define OFF_WIHT  ((size_t)50331648)                 // f16 packed, 32768*384    (25,165,824 B)
#define OFF_WFFN  ((size_t)75497472)                 // f16 packed, 128*16384    ( 4,194,304 B)
#define OFF_WHHT  ((size_t)79691776)                 // f16 packed, 128*384      (    98,304 B)
#define OFF_WSP   ((size_t)79790080)                 // f16 packed, 128*128      (    32,768 B)
#define OFF_HT    ((size_t)79822848)                 // f16, 32*128              (     8,192 B)
#define OFF_GI    ((size_t)79831040)                 // f32, 384*384             (   589,824 B)
#define OFF_ACC   ((size_t)80420864)                 // f32 x2 (mse, bce sums)

// ---------------------------------------------------------------------------
// WMMA fragment helpers (v_wmma_f32_16x16x32_f16; this code uses contiguous k = 16h + s in both A and B)
// ---------------------------------------------------------------------------

// A is 16x32 (MxK) row-major with stride lda (halves). Lane holds
// row=lane%16, K in {h*16..h*16+15}; both runs contiguous & 16B
// aligned -> two b128 loads (global or ds depending on pointer origin).
__device__ inline v16h load_a_frag(const half_t* base, int lda) {
    const int lane = threadIdx.x & 31;
    const int r = lane & 15;
    const int h = lane >> 4;
    const half_t* p = base + (size_t)r * lda + h * 16;
    v8h a0 = *(const v8h*)(p);
    v8h a1 = *(const v8h*)(p + 8);
    return __builtin_shufflevector(a0, a1, 0, 1, 2, 3, 4, 5, 6, 7,
                                   8, 9, 10, 11, 12, 13, 14, 15);
}

// B tiles are pre-packed: tile = 32 lanes x 16 halves, lane-contiguous.
__device__ inline v16h load_b_packed(const half_t* tile_base) {
    const int lane = threadIdx.x & 31;
    const half_t* p = tile_base + lane * 16;
    v8h b0 = *(const v8h*)(p);
    v8h b1 = *(const v8h*)(p + 8);
    return __builtin_shufflevector(b0, b1, 0, 1, 2, 3, 4, 5, 6, 7,
                                   8, 9, 10, 11, 12, 13, 14, 15);
}

__device__ inline v8f wmma_f16(v16h a, v16h b, v8f c) {
    return __builtin_amdgcn_wmma_f32_16x16x32_f16(false, a, false, b, (short)0, c,
                                                  false, false);
}

// ---------------------------------------------------------------------------
// CDNA5 async global->LDS copy (ASYNCcnt-tracked). 16 bytes per lane per call.
// Builtin signature (from clang): (v4i as1*, v4i as3*, imm offset, imm cpol).
// Falls back to register-staged LDS stores if the builtin is unavailable.
// ---------------------------------------------------------------------------
#if __has_builtin(__builtin_amdgcn_global_load_async_to_lds_b128)
#define HAVE_ASYNC_LDS 1
typedef __attribute__((address_space(1))) v4i gv4i;
typedef __attribute__((address_space(3))) v4i lv4i;
#else
#define HAVE_ASYNC_LDS 0
#endif

__device__ inline void async_cp16B(const half_t* g, half_t* l) {
#if HAVE_ASYNC_LDS
    __builtin_amdgcn_global_load_async_to_lds_b128((gv4i*)g, (lv4i*)l, 0, 0);
#else
    *(v8h*)l = *(const v8h*)g;
#endif
}

__device__ inline void wait_async_done() {
#if HAVE_ASYNC_LDS
#if __has_builtin(__builtin_amdgcn_s_wait_asynccnt)
    __builtin_amdgcn_s_wait_asynccnt(0);
#else
    asm volatile("s_wait_asynccnt 0x0" ::: "memory");
#endif
#endif
}

__device__ inline float sigm(float x) { return 1.0f / (1.0f + __expf(-x)); }

// 3x3 SAME cross-correlation, 1->2 channels, on a 128x128 image (adds bias)
__device__ inline void cond_conv(const float* img, int r, int c,
                                 const float* fw, float fb0, float fb1,
                                 float& h0, float& h1) {
    float s0 = 0.f, s1 = 0.f;
#pragma unroll
    for (int dr = -1; dr <= 1; ++dr) {
        int rr = r + dr;
        if (rr < 0 || rr >= NN) continue;
#pragma unroll
        for (int dc = -1; dc <= 1; ++dc) {
            int cc = c + dc;
            if (cc < 0 || cc >= NN) continue;
            float v = img[rr * NN + cc];
            int wi = (dr + 1) * 3 + (dc + 1);
            s0 += v * fw[wi];
            s1 += v * fw[9 + wi];
        }
    }
    h0 = s0 + fb0;
    h1 = s1 + fb1;
}

// ---------------------------------------------------------------------------
// K0: zero loss accumulators
// ---------------------------------------------------------------------------
__global__ void k_zero_acc(float* acc) {
    if (threadIdx.x < 2) acc[threadIdx.x] = 0.0f;
}

// ---------------------------------------------------------------------------
// K1: pack weights (f32 -> f16) into WMMA-B fragment layout.
//     B logical matrix is K x Nc. srcT=0: src[k*Nc+o]; srcT=1: src[o*K+k].
//     dst[((kt*ntiles+nt)*32 + lane)*16 + e] = B[kt*32 + h*16 + e][nt*16 + n]
// ---------------------------------------------------------------------------
__global__ void k_pack_b(const float* __restrict__ src, half_t* __restrict__ dst,
                         int K, int Nc, int srcT) {
    size_t i = (size_t)blockIdx.x * blockDim.x + threadIdx.x;
    size_t total = (size_t)K * Nc;
    if (i >= total) return;
    int e = (int)(i & 15);
    int lane = (int)((i >> 4) & 31);
    size_t tile = i >> 9;
    int ntiles = Nc / 16;
    int nt = (int)(tile % ntiles);
    int kt = (int)(tile / ntiles);
    int n = lane & 15;
    int h = lane >> 4;
    int k = kt * 32 + h * 16 + e;
    int o = nt * 16 + n;
    float v = srcT ? src[(size_t)o * K + k] : src[(size_t)k * Nc + o];
    dst[i] = (half_t)v;
}

// ---------------------------------------------------------------------------
// K2: conditional flow forward + sigmoid -> HR_ge (f16) ; fused MSE/BCE sums.
// ---------------------------------------------------------------------------
__global__ void k_flow_fwd(const float* __restrict__ HR, const float* __restrict__ LR,
                           const float* __restrict__ flow_w, const float* __restrict__ flow_b,
                           half_t* __restrict__ hge, float* __restrict__ acc) {
    const int bw = blockIdx.x;
    const int b = bw / WW, w = bw % WW;
    const int wc = (w > 0) ? (w - 1) : 0;
    const float* cond  = LR + ((size_t)b * WW + wc) * PIX;
    const float* hr_in = LR + ((size_t)b * WW + w) * PIX;
    const float* HRb   = HR + (size_t)bw * 2 * PIX;
    half_t* ge         = hge + (size_t)bw * 2 * PIX;

    float fw[18];
#pragma unroll
    for (int i = 0; i < 18; ++i) fw[i] = flow_w[i];
    const float fb0 = flow_b[0], fb1 = flow_b[1];

    float msesum = 0.f, bcesum = 0.f;
    for (int p = threadIdx.x; p < PIX; p += 256) {
        int r = p >> 7, c = p & 127;
        float h0, h1;
        cond_conv(cond, r, c, fw, fb0, fb1, h0, h1);
        float log_s = tanhf(h0);
        float t     = h1;
        float z     = hr_in[p] * __expf(log_s) + t;
        float g0 = sigm(z);   // channel 0 -> rows [0, N)
        float g1 = sigm(t);   // channel 1 -> rows [N, 2N)
        ge[p]       = (half_t)g0;
        ge[PIX + p] = (half_t)g1;
        {
            float x = HRb[p];
            float d = x - g0;
            msesum += d * d;
            float lp = fmaxf(logf(x), -100.f);
            float l1 = fmaxf(logf(1.f - x), -100.f);
            bcesum += -(g0 * lp + (1.f - g0) * l1);
        }
        {
            float x = HRb[PIX + p];
            float d = x - g1;
            msesum += d * d;
            float lp = fmaxf(logf(x), -100.f);
            float l1 = fmaxf(logf(1.f - x), -100.f);
            bcesum += -(g1 * lp + (1.f - g1) * l1);
        }
    }
    __shared__ float red[256];
    red[threadIdx.x] = msesum;
    __syncthreads();
    for (int s = 128; s > 0; s >>= 1) {
        if (threadIdx.x < s) red[threadIdx.x] += red[threadIdx.x + s];
        __syncthreads();
    }
    if (threadIdx.x == 0) atomicAdd(&acc[0], red[0]);
    __syncthreads();
    red[threadIdx.x] = bcesum;
    __syncthreads();
    for (int s = 128; s > 0; s >>= 1) {
        if (threadIdx.x < s) red[threadIdx.x] += red[threadIdx.x + s];
        __syncthreads();
    }
    if (threadIdx.x == 0) atomicAdd(&acc[1], red[0]);
}

// ---------------------------------------------------------------------------
// K3: spconv  te = relu(HR_ge @ W_sp + b_sp), f16 out.
//     grid = (16 mtiles, 384 bw) x 256 threads (8 waves = 8 ntiles).
// ---------------------------------------------------------------------------
__global__ void k_spconv(const half_t* __restrict__ hge, const half_t* __restrict__ WspP,
                         const float* __restrict__ b_sp, half_t* __restrict__ te) {
    const int bw = blockIdx.y;
    const int mt = blockIdx.x;
    const int nt = threadIdx.x >> 5;   // wave id -> ntile (8 ntiles)
    const half_t* A0 = hge + (size_t)bw * NODES * NN + (size_t)mt * 16 * NN;
    v8f c = {};
#pragma unroll
    for (int kt = 0; kt < NN / 32; ++kt) {
        v16h a = load_a_frag(A0 + kt * 32, NN);
        v16h bm = load_b_packed(WspP + (size_t)(kt * 8 + nt) * TILE_HALFS);
        c = wmma_f16(a, bm, c);
    }
    const int lane = threadIdx.x & 31;
    const int n = lane & 15, hh = lane >> 4;
    const int col = nt * 16 + n;
    const float bias = b_sp[col];
    half_t* out = te + (size_t)bw * NODES * HH + (size_t)mt * 16 * HH;
#pragma unroll
    for (int j = 0; j < 8; ++j) {
        float v = c[j] + bias;
        v = v > 0.f ? v : 0.f;
        out[(size_t)(hh * 8 + j) * HH + col] = (half_t)v;
    }
}

// ---------------------------------------------------------------------------
// K4: GI = X @ W_ihT + b_ih   (384 x 32768) x (32768 x 384) -> f32.
//     grid = 12 blocks (32 rows each) x 256 threads (8 waves).
//     Double-buffered async global->LDS staging of the 25MB weight stream and
//     the shared A strip; each wave computes 2 mtiles x 3 ntiles per A pair.
// ---------------------------------------------------------------------------
__device__ inline void gi_stage(const half_t* __restrict__ Ag,
                                const half_t* __restrict__ WihP, int kt,
                                half_t* Asb, half_t* Bsb) {
    const int wave = threadIdx.x >> 5;
    const int lane = threadIdx.x & 31;
    // B: this wave's 3 tiles (1KB each -> 2 wave-wide 512B copies per tile)
    const half_t* brow = WihP + (size_t)kt * NTI * TILE_HALFS;
#pragma unroll
    for (int q = 0; q < 3; ++q) {
        const int nt = wave + q * 8;
        const half_t* g = brow + (size_t)nt * TILE_HALFS;
        half_t* l = Bsb + nt * TILE_HALFS;
        async_cp16B(g + lane * 8, l + lane * 8);
        async_cp16B(g + 256 + lane * 8, l + 256 + lane * 8);
    }
    // A: 2 fragments (16x32 rows, strided in global) staged row-major in LDS.
    // 4 x 512B wave-copies; waves 0..3 take one each.
    if (wave < 4) {
        const int f = wave >> 1, p = wave & 1;
        const int row = p * 8 + (lane >> 2);
        const int colh = (lane & 3) * 8;
        const half_t* g = Ag + ((size_t)(f * 16 + row)) * IN_FEAT +
                          (size_t)kt * 32 + colh;
        half_t* l = Asb + f * TILE_HALFS + row * 32 + colh;
        async_cp16B(g, l);
    }
}

__global__ void k_gi_gemm(const half_t* __restrict__ X, const half_t* __restrict__ WihP,
                          const float* __restrict__ b_ih, float* __restrict__ GI) {
    __shared__ __align__(16) half_t Asm[2][2 * TILE_HALFS];     //  2 x 2KB
    __shared__ __align__(16) half_t Bsm[2][NTI * TILE_HALFS];   //  2 x 24KB
    const int wave = threadIdx.x >> 5;
    const half_t* Ag = X + (size_t)blockIdx.x * 32 * IN_FEAT;

    v8f c00 = {}, c01 = {}, c02 = {};
    v8f c10 = {}, c11 = {}, c12 = {};

    gi_stage(Ag, WihP, 0, Asm[0], Bsm[0]);

    for (int kt = 0; kt < KTI; ++kt) {
        const int cur = kt & 1;
        wait_async_done();      // my stage(kt) copies have landed
        __syncthreads();        // everyone's have; prior reads of buf[1-cur] done
        if (kt + 1 < KTI) gi_stage(Ag, WihP, kt + 1, Asm[cur ^ 1], Bsm[cur ^ 1]);

        v16h a0 = load_a_frag(Asm[cur], 32);
        v16h a1 = load_a_frag(Asm[cur] + TILE_HALFS, 32);
        {
            v16h b = load_b_packed(Bsm[cur] + (size_t)(wave)      * TILE_HALFS);
            c00 = wmma_f16(a0, b, c00);
            c10 = wmma_f16(a1, b, c10);
        }
        {
            v16h b = load_b_packed(Bsm[cur] + (size_t)(wave + 8)  * TILE_HALFS);
            c01 = wmma_f16(a0, b, c01);
            c11 = wmma_f16(a1, b, c11);
        }
        {
            v16h b = load_b_packed(Bsm[cur] + (size_t)(wave + 16) * TILE_HALFS);
            c02 = wmma_f16(a0, b, c02);
            c12 = wmma_f16(a1, b, c12);
        }
    }

    const int lane = threadIdx.x & 31;
    const int n = lane & 15, hh = lane >> 4;
#pragma unroll
    for (int m = 0; m < 2; ++m) {
#pragma unroll
        for (int q = 0; q < 3; ++q) {
            const int nt = wave + q * 8;
            const int col = nt * 16 + n;
            const float bias = b_ih[col];
            const v8f& c = (m == 0) ? ((q == 0) ? c00 : (q == 1) ? c01 : c02)
                                    : ((q == 0) ? c10 : (q == 1) ? c11 : c12);
            const size_t row0 = (size_t)blockIdx.x * 32 + m * 16 + hh * 8;
#pragma unroll
            for (int j = 0; j < 8; ++j)
                GI[(row0 + j) * GATES + col] = c[j] + bias;
        }
    }
}

// ---------------------------------------------------------------------------
// K5: GRU recurrence over W=12 steps; single workgroup, h in LDS (f16).
//     gh = h @ W_hhT + b_hh via WMMA (2x24 tiles, K=4), then gate math.
// ---------------------------------------------------------------------------
__global__ void k_gru(const float* __restrict__ GI, const half_t* __restrict__ WhhP,
                      const float* __restrict__ b_hh, half_t* __restrict__ hT) {
    __shared__ __align__(16) half_t h_sm[BB * HH];   // 32 x 128
    __shared__ float gh_sm[BB * GATES];              // 32 x 384
    const int tid = threadIdx.x;
    const int wave = tid >> 5;

    for (int i = tid; i < BB * HH; i += 256) h_sm[i] = (half_t)0.0f;
    __syncthreads();

    for (int t = 0; t < WW; ++t) {
        // --- gh GEMM: 48 tiles, 6 per wave ---
#pragma unroll
        for (int tt = 0; tt < 6; ++tt) {
            const int tile = wave * 6 + tt;
            const int mt = tile / NTI;
            const int nt = tile % NTI;
            const half_t* A0 = &h_sm[mt * 16 * HH];
            v8f c = {};
#pragma unroll
            for (int kt = 0; kt < HH / 32; ++kt) {
                v16h a = load_a_frag(A0 + kt * 32, HH);
                v16h bm = load_b_packed(WhhP + (size_t)(kt * NTI + nt) * TILE_HALFS);
                c = wmma_f16(a, bm, c);
            }
            const int lane = tid & 31;
            const int n = lane & 15, hh2 = lane >> 4;
            const int col = nt * 16 + n;
            const float bias = b_hh[col];
#pragma unroll
            for (int j = 0; j < 8; ++j)
                gh_sm[(mt * 16 + hh2 * 8 + j) * GATES + col] = c[j] + bias;
        }
        __syncthreads();

        // --- gate math: 32*128 elements ---
        for (int i = tid; i < BB * HH; i += 256) {
            const int b = i >> 7, j = i & 127;
            const float* gi = GI + ((size_t)b * WW + t) * GATES;
            float gir = gi[j], giz = gi[HH + j], gin = gi[2 * HH + j];
            float ghr = gh_sm[b * GATES + j];
            float ghz = gh_sm[b * GATES + HH + j];
            float ghn = gh_sm[b * GATES + 2 * HH + j];
            float r = sigm(gir + ghr);
            float z = sigm(giz + ghz);
            float nn = tanhf(gin + r * ghn);
            float hp = (float)h_sm[i];
            h_sm[i] = (half_t)((1.f - z) * nn + z * hp);
        }
        __syncthreads();
    }
    for (int i = tid; i < BB * HH; i += 256) hT[i] = h_sm[i];
}

// ---------------------------------------------------------------------------
// K6: FFN  out = sigmoid(hT @ W_ffn + b_ffn)  (32 x 128)x(128 x 16384)
// ---------------------------------------------------------------------------
__global__ void k_ffn(const half_t* __restrict__ hT, const half_t* __restrict__ WffnP,
                      const float* __restrict__ b_ffn, float* __restrict__ outD) {
    const int mt = blockIdx.y;
    const int nt = blockIdx.x * 8 + (threadIdx.x >> 5);
    const int NT = OUT_FFN / 16;       // 1024 ntiles
    const half_t* A0 = hT + (size_t)mt * 16 * HH;
    v8f c = {};
#pragma unroll
    for (int kt = 0; kt < HH / 32; ++kt) {
        v16h a = load_a_frag(A0 + kt * 32, HH);
        v16h bm = load_b_packed(WffnP + (size_t)(kt * NT + nt) * TILE_HALFS);
        c = wmma_f16(a, bm, c);
    }
    const int lane = threadIdx.x & 31;
    const int n = lane & 15, hh = lane >> 4;
    const int col = nt * 16 + n;
    const float bias = b_ffn[col];
#pragma unroll
    for (int j = 0; j < 8; ++j) {
        float v = sigm(c[j] + bias);
        if (isnan(v)) v = 0.f;
        outD[((size_t)mt * 16 + hh * 8 + j) * OUT_FFN + col] = v;
    }
}

// ---------------------------------------------------------------------------
// K7: reverse flow conditioned on LR[:, -1] -> outputLR at d_out[524288..]
// ---------------------------------------------------------------------------
__global__ void k_rev_flow(const float* __restrict__ LR, const float* __restrict__ flow_w,
                           const float* __restrict__ flow_b, float* __restrict__ outD) {
    const int b = blockIdx.x;
    const float* cond = LR + ((size_t)b * WW + (WW - 1)) * PIX;
    const float* z2   = outD + (size_t)b * PIX;
    float* olr        = outD + (size_t)BB * PIX + (size_t)b * PIX;

    float fw[18];
#pragma unroll
    for (int i = 0; i < 18; ++i) fw[i] = flow_w[i];
    const float fb0 = flow_b[0], fb1 = flow_b[1];

    for (int p = threadIdx.x; p < PIX; p += 256) {
        int r = p >> 7, c = p & 127;
        float h0, h1;
        cond_conv(cond, r, c, fw, fb0, fb1, h0, h1);
        float log_s = tanhf(h0);
        float t2 = h1;
        float hr = (z2[p] - t2) * __expf(-log_s);
        olr[p] = sigm(hr);
    }
}

// ---------------------------------------------------------------------------
// K8: finalize scalar loss at d_out[1048576]
// ---------------------------------------------------------------------------
__global__ void k_finalize(const float* __restrict__ acc, float* __restrict__ outD) {
    if (threadIdx.x == 0 && blockIdx.x == 0) {
        float mse = acc[0] / LOSS_CNT;
        float bce = acc[1] / LOSS_CNT;
        outD[(size_t)2 * BB * PIX] = mse + bce / (float)BW;
    }
}

// ---------------------------------------------------------------------------
extern "C" void kernel_launch(void* const* d_in, const int* in_sizes, int n_in,
                              void* d_out, int out_size, void* d_ws, size_t ws_size,
                              hipStream_t stream) {
    const float* HR     = (const float*)d_in[0];
    const float* LR     = (const float*)d_in[1];
    const float* flow_w = (const float*)d_in[2];
    const float* flow_b = (const float*)d_in[3];
    const float* W_sp   = (const float*)d_in[4];
    const float* b_sp   = (const float*)d_in[5];
    const float* W_ih   = (const float*)d_in[6];
    const float* W_hh   = (const float*)d_in[7];
    const float* b_ih   = (const float*)d_in[8];
    const float* b_hh   = (const float*)d_in[9];
    const float* W_ffn  = (const float*)d_in[10];
    const float* b_ffn  = (const float*)d_in[11];
    float* out = (float*)d_out;

    char* ws = (char*)d_ws;
    half_t* hge   = (half_t*)(ws + OFF_HGE);
    half_t* te    = (half_t*)(ws + OFF_TE);
    half_t* WihP  = (half_t*)(ws + OFF_WIHT);
    half_t* WffnP = (half_t*)(ws + OFF_WFFN);
    half_t* WhhP  = (half_t*)(ws + OFF_WHHT);
    half_t* WspP  = (half_t*)(ws + OFF_WSP);
    half_t* hT    = (half_t*)(ws + OFF_HT);
    float*  GI    = (float*)(ws + OFF_GI);
    float*  acc   = (float*)(ws + OFF_ACC);

    // accumulators must be re-zeroed every call (graph replay safe)
    k_zero_acc<<<1, 32, 0, stream>>>(acc);

    // weight packing (f32 -> f16, fragment-native B layout)
    k_pack_b<<<(NN * HH + 255) / 256, 256, 0, stream>>>(W_sp, WspP, NN, HH, 0);
    k_pack_b<<<(HH * OUT_FFN + 255) / 256, 256, 0, stream>>>(W_ffn, WffnP, HH, OUT_FFN, 0);
    k_pack_b<<<(HH * GATES + 255) / 256, 256, 0, stream>>>(W_hh, WhhP, HH, GATES, 1);
    {
        size_t n = (size_t)IN_FEAT * GATES;
        k_pack_b<<<(unsigned)((n + 255) / 256), 256, 0, stream>>>(W_ih, WihP, IN_FEAT, GATES, 1);
    }

    // flow forward + fused loss
    k_flow_fwd<<<BW, 256, 0, stream>>>(HR, LR, flow_w, flow_b, hge, acc);

    // spconv: (B*W*2N x N) @ (N x H) with relu -> te (f16)
    k_spconv<<<dim3(NODES / 16, BW), 256, 0, stream>>>(hge, WspP, b_sp, te);

    // GI = X @ W_ihT + b_ih : dominant GEMM, async double-buffered staging
    k_gi_gemm<<<BW / 32, 256, 0, stream>>>(te, WihP, b_ih, GI);

    // sequential GRU recurrence (tiny per-step GEMM, single persistent WG)
    k_gru<<<1, 256, 0, stream>>>(GI, WhhP, b_hh, hT);

    // FFN + sigmoid -> output (d_out[0 .. 524287])
    k_ffn<<<dim3(OUT_FFN / 128, BB / 16), 256, 0, stream>>>(hT, WffnP, b_ffn, out);

    // reverse flow -> outputLR (d_out[524288 .. 1048575])
    k_rev_flow<<<BB, 256, 0, stream>>>(LR, flow_w, flow_b, out);

    // scalar loss -> d_out[1048576]
    k_finalize<<<1, 32, 0, stream>>>(acc, out);
}
